// MultiHeadAttention_58042188038440
// MI455X (gfx1250) — matrix operand order repaired and checked by reading, not compiled, not run on hardware
//
#include <hip/hip_runtime.h>
#include <hip/hip_bf16.h>

#define HIDDEN 1024
#define SEQ    2048
#define NHEAD  16
#define DHEAD  64

typedef __attribute__((ext_vector_type(16))) __bf16 v16bf;
typedef __attribute__((ext_vector_type(8)))  float  v8f;

union Frag16 { v16bf v; uint4 q[2]; };

__device__ __forceinline__ v8f zero8() {
  v8f z = {0.f,0.f,0.f,0.f,0.f,0.f,0.f,0.f};
  return z;
}

__device__ __forceinline__ v8f wmma_bf16(v16bf a, v16bf b, v8f c) {
  // D(16x16,f32) = A(16x32,bf16) x B(32x16,bf16) + C
  return __builtin_amdgcn_wmma_f32_16x16x32_bf16(
      /*neg_a=*/false, a, /*neg_b=*/false, b,
      /*c_mod=*/(short)0, c, /*reuse_a=*/false, /*reuse_b=*/false);
}

// A fragment (16x32 bf16): rows row0..+15, cols col0..+31 of row-major src.
// ISA layout: lane L -> M=L&15; lanes<16 hold K=0..7,16..23; lanes>=16 hold K=8..15,24..31.
__device__ __forceinline__ v16bf load_frag_a(const __bf16* src, int row0, int col0, int ld) {
  int lane = threadIdx.x & 31;
  const __bf16* p = src + (size_t)(row0 + (lane & 15)) * (size_t)ld + col0 + (lane >> 4) * 8;
  Frag16 f;
  f.q[0] = *(const uint4*)(p);        // K run 0..7  (of this half)
  f.q[1] = *(const uint4*)(p + 16);   // K run 16..23 (of this half)
  return f.v;
}

// B fragment (32x16 bf16) where src is row-major holding B^T:
// B[K][N] = src[nrow0+N][k0+K]. Lane L -> N=L&15; lanes<16 K=0..7,16..23; lanes>=16 K=8..15,24..31.
__device__ __forceinline__ v16bf load_frag_bt(const __bf16* src, int nrow0, int k0, int ld) {
  int lane = threadIdx.x & 31;
  const __bf16* p = src + (size_t)(nrow0 + (lane & 15)) * (size_t)ld + k0 + (lane >> 4) * 8;
  Frag16 f;
  f.q[0] = *(const uint4*)(p);
  f.q[1] = *(const uint4*)(p + 16);
  return f.v;
}

// ---------------- LayerNorm (fp32 in) -> bf16 normalized activations ----------------
__global__ void __launch_bounds__(256) ln_kernel(const float* __restrict__ x,
                                                 const float* __restrict__ g,
                                                 const float* __restrict__ b,
                                                 __bf16* __restrict__ xn) {
  int row = blockIdx.x;                       // 0..4095
  const float4* xr = (const float4*)(x + (size_t)row * HIDDEN);
  float4 v = xr[threadIdx.x];                 // 256 threads * 4 = 1024
  float s  = v.x + v.y + v.z + v.w;
  float s2 = v.x*v.x + v.y*v.y + v.z*v.z + v.w*v.w;
  #pragma unroll
  for (int m = 1; m <= 16; m <<= 1) {
    s  += __shfl_xor(s,  m, 32);
    s2 += __shfl_xor(s2, m, 32);
  }
  __shared__ float ss[8], ss2[8];
  int wid = threadIdx.x >> 5, lane = threadIdx.x & 31;
  if (lane == 0) { ss[wid] = s; ss2[wid] = s2; }
  __syncthreads();
  float tot = 0.f, tot2 = 0.f;
  #pragma unroll
  for (int i = 0; i < 8; ++i) { tot += ss[i]; tot2 += ss2[i]; }
  float mu   = tot * (1.0f / HIDDEN);
  float var  = tot2 * (1.0f / HIDDEN) - mu * mu;
  float rstd = rsqrtf(var + 1e-5f);
  float4 gg = ((const float4*)g)[threadIdx.x];
  float4 bb = ((const float4*)b)[threadIdx.x];
  __bf16* o = xn + (size_t)row * HIDDEN + threadIdx.x * 4;
  o[0] = (__bf16)((v.x - mu) * rstd * gg.x + bb.x);
  o[1] = (__bf16)((v.y - mu) * rstd * gg.y + bb.y);
  o[2] = (__bf16)((v.z - mu) * rstd * gg.z + bb.z);
  o[3] = (__bf16)((v.w - mu) * rstd * gg.w + bb.w);
}

// ---------------- fp32 -> bf16 weight conversion (z selects which weight) ----------------
__global__ void __launch_bounds__(256) cvt_kernel(const float* __restrict__ w0,
                                                  const float* __restrict__ w1,
                                                  const float* __restrict__ w2,
                                                  const float* __restrict__ w3,
                                                  __bf16* __restrict__ dst0) {
  const float* src;
  __bf16* dst = dst0 + (size_t)blockIdx.z * (HIDDEN * HIDDEN);
  if (blockIdx.z == 0)      src = w0;
  else if (blockIdx.z == 1) src = w1;
  else if (blockIdx.z == 2) src = w2;
  else                      src = w3;
  int i = blockIdx.x * 256 + threadIdx.x;     // grid.x * 256 * 4 == HIDDEN*HIDDEN
  float4 v = ((const float4*)src)[i];
  __bf16* o = dst + (size_t)i * 4;
  o[0] = (__bf16)v.x; o[1] = (__bf16)v.y; o[2] = (__bf16)v.z; o[3] = (__bf16)v.w;
}

// ---------------- QKV projection: out = (xn @ W^T + bias) * scale, bf16 out ----------------
// block = 256 threads = 8 waves in 4x2; wave tile 32(M)x64(N); block tile 128x128.
__global__ void __launch_bounds__(256) qkv_gemm(const __bf16* __restrict__ xn,
                                                const __bf16* __restrict__ wq,
                                                const __bf16* __restrict__ wk,
                                                const __bf16* __restrict__ wv,
                                                const float* __restrict__ bq,
                                                const float* __restrict__ bk,
                                                const float* __restrict__ bv,
                                                __bf16* __restrict__ qo,
                                                __bf16* __restrict__ ko,
                                                __bf16* __restrict__ vo) {
  const __bf16* W; const float* bias; __bf16* out; float scale = 1.0f;
  if (blockIdx.z == 0)      { W = wq; bias = bq; out = qo; scale = 0.125f; } // fold 1/sqrt(64)
  else if (blockIdx.z == 1) { W = wk; bias = bk; out = ko; }
  else                      { W = wv; bias = bv; out = vo; }

  int wid = threadIdx.x >> 5;
  int wm = wid >> 1, wn = wid & 1;               // 4 x 2 wave grid
  int m0 = blockIdx.x * 128 + wm * 32;
  int n0 = blockIdx.y * 128 + wn * 64;

  v8f acc[2][4];
  #pragma unroll
  for (int mi = 0; mi < 2; ++mi)
    #pragma unroll
    for (int ni = 0; ni < 4; ++ni) acc[mi][ni] = zero8();

  for (int k0 = 0; k0 < HIDDEN; k0 += 32) {
    v16bf a0 = load_frag_a(xn, m0,      k0, HIDDEN);
    v16bf a1 = load_frag_a(xn, m0 + 16, k0, HIDDEN);
    #pragma unroll
    for (int ni = 0; ni < 4; ++ni) {
      v16bf bfr = load_frag_bt(W, n0 + ni * 16, k0, HIDDEN);  // W[n][k] is B^T
      acc[0][ni] = wmma_bf16(a0, bfr, acc[0][ni]);
      acc[1][ni] = wmma_bf16(a1, bfr, acc[1][ni]);
    }
  }

  int lane = threadIdx.x & 31, half = lane >> 4, nc = lane & 15;
  #pragma unroll
  for (int mi = 0; mi < 2; ++mi)
    #pragma unroll
    for (int ni = 0; ni < 4; ++ni)
      #pragma unroll
      for (int r = 0; r < 8; ++r) {
        int m = m0 + mi * 16 + half * 8 + r;
        int n = n0 + ni * 16 + nc;
        out[(size_t)m * HIDDEN + n] = (__bf16)((acc[mi][ni][r] + bias[n]) * scale);
      }
}

// ---------------- Flash attention: per block one (batch, head, 256 q rows) ----------------
// 8 waves, each owns 32 q rows; stream K/V in 64-key tiles through LDS with online softmax.
__global__ void __launch_bounds__(256) attn_kernel(const __bf16* __restrict__ q,
                                                   const __bf16* __restrict__ k,
                                                   const __bf16* __restrict__ v,
                                                   __bf16* __restrict__ ctx) {
  __shared__ __bf16 Kt[64][72];       // K tile, row-major [key][d]
  __shared__ __bf16 Vt[64][72];       // V tile transposed [d][key]
  __shared__ __bf16 Pb[8][32][72];    // per-wave probability tile [qrow][key]

  int head = blockIdx.y, batch = blockIdx.z;
  int hd = head * DHEAD;
  int rowBase = batch * SEQ;
  int tid = threadIdx.x;
  int wid = tid >> 5, lane = tid & 31, half = lane >> 4, nc = lane & 15;
  int q0 = blockIdx.x * 256 + wid * 32;            // local q row of this wave

  // Q A-fragments: resident for whole kernel (2 M-tiles x 2 K-steps over d=64)
  v16bf aq[2][2];
  #pragma unroll
  for (int mi = 0; mi < 2; ++mi)
    #pragma unroll
    for (int kk = 0; kk < 2; ++kk)
      aq[mi][kk] = load_frag_a(q, rowBase + q0 + mi * 16, hd + kk * 32, HIDDEN);

  v8f o[2][4];
  float m_st[2][8], l_st[2][8];
  #pragma unroll
  for (int mi = 0; mi < 2; ++mi) {
    #pragma unroll
    for (int nj = 0; nj < 4; ++nj) o[mi][nj] = zero8();
    #pragma unroll
    for (int r = 0; r < 8; ++r) { m_st[mi][r] = -INFINITY; l_st[mi][r] = 0.f; }
  }

  for (int t = 0; t < SEQ; t += 64) {
    __syncthreads();  // previous tile fully consumed before overwrite
    // cooperative K tile load: 64 rows x 128B, uint4 chunks
    for (int i = tid; i < 512; i += 256) {
      int key = i >> 3, c = i & 7;
      *(uint4*)&Kt[key][c * 8] =
          *(const uint4*)(k + (size_t)(rowBase + t + key) * HIDDEN + hd + c * 8);
    }
    // cooperative V tile load, transposed into Vt[d][key]
    for (int i = tid; i < 4096; i += 256) {
      int key = i >> 6, d = i & 63;
      Vt[d][key] = v[(size_t)(rowBase + t + key) * HIDDEN + hd + d];
    }
    __syncthreads();

    // ---- scores S = Q K^T (scale pre-folded into Q) ----
    v8f s[2][4];
    #pragma unroll
    for (int mi = 0; mi < 2; ++mi)
      #pragma unroll
      for (int nj = 0; nj < 4; ++nj) s[mi][nj] = zero8();
    #pragma unroll
    for (int kk = 0; kk < 2; ++kk)
      #pragma unroll
      for (int nj = 0; nj < 4; ++nj) {
        v16bf bk = load_frag_bt(&Kt[0][0], nj * 16, kk * 32, 72);
        s[0][nj] = wmma_bf16(aq[0][kk], bk, s[0][nj]);
        s[1][nj] = wmma_bf16(aq[1][kk], bk, s[1][nj]);
      }

    // ---- online softmax per M-tile ----
    #pragma unroll
    for (int mi = 0; mi < 2; ++mi) {
      float mnew[8], alpha[8];
      #pragma unroll
      for (int r = 0; r < 8; ++r) {
        float mx = fmaxf(fmaxf(s[mi][0][r], s[mi][1][r]), fmaxf(s[mi][2][r], s[mi][3][r]));
        mx = fmaxf(mx, __shfl_xor(mx, 1, 32));
        mx = fmaxf(mx, __shfl_xor(mx, 2, 32));
        mx = fmaxf(mx, __shfl_xor(mx, 4, 32));
        mx = fmaxf(mx, __shfl_xor(mx, 8, 32));   // stays within each 16-lane half = one N-stripe
        float mn = fmaxf(m_st[mi][r], mx);
        alpha[r] = __expf(m_st[mi][r] - mn);
        m_st[mi][r] = mn;
        mnew[r] = mn;
      }
      #pragma unroll
      for (int r = 0; r < 8; ++r) {
        float rs = 0.f;
        #pragma unroll
        for (int nj = 0; nj < 4; ++nj) {
          float p = __expf(s[mi][nj][r] - mnew[r]);
          s[mi][nj][r] = p;
          rs += p;
        }
        rs += __shfl_xor(rs, 1, 32);
        rs += __shfl_xor(rs, 2, 32);
        rs += __shfl_xor(rs, 4, 32);
        rs += __shfl_xor(rs, 8, 32);
        l_st[mi][r] = l_st[mi][r] * alpha[r] + rs;
        #pragma unroll
        for (int nj = 0; nj < 4; ++nj) o[mi][nj][r] *= alpha[r];
      }
      // spill P (C-frag layout) to LDS for re-read in A-frag layout
      #pragma unroll
      for (int nj = 0; nj < 4; ++nj)
        #pragma unroll
        for (int r = 0; r < 8; ++r)
          Pb[wid][mi * 16 + half * 8 + r][nj * 16 + nc] = (__bf16)s[mi][nj][r];
    }
    asm volatile("s_wait_dscnt 0" ::: "memory");  // same-wave P writes -> reads

    // ---- O += P @ V ----
    #pragma unroll
    for (int kk = 0; kk < 2; ++kk) {
      v16bf ap0 = load_frag_a(&Pb[wid][0][0], 0,  kk * 32, 72);
      v16bf ap1 = load_frag_a(&Pb[wid][0][0], 16, kk * 32, 72);
      #pragma unroll
      for (int nj = 0; nj < 4; ++nj) {
        v16bf bv = load_frag_bt(&Vt[0][0], nj * 16, kk * 32, 72);  // Vt[d][key] = B^T
        o[0][nj] = wmma_bf16(ap0, bv, o[0][nj]);
        o[1][nj] = wmma_bf16(ap1, bv, o[1][nj]);
      }
    }
  }

  // epilogue: normalize and store ctx (bf16)
  #pragma unroll
  for (int mi = 0; mi < 2; ++mi)
    #pragma unroll
    for (int r = 0; r < 8; ++r) {
      float inv = 1.0f / l_st[mi][r];
      int m = rowBase + q0 + mi * 16 + half * 8 + r;
      #pragma unroll
      for (int nj = 0; nj < 4; ++nj)
        ctx[(size_t)m * HIDDEN + hd + nj * 16 + nc] = (__bf16)(o[mi][nj][r] * inv);
    }
}

// ---------------- Output projection + residual: out = x + ctx @ Wo^T + bo (fp32) ----------------
// Same 32x64 wave tile / 128x128 block tile as qkv_gemm.
__global__ void __launch_bounds__(256) oproj_gemm(const __bf16* __restrict__ ctx,
                                                  const __bf16* __restrict__ wo,
                                                  const float* __restrict__ bo,
                                                  const float* __restrict__ x,
                                                  float* __restrict__ out) {
  int wid = threadIdx.x >> 5;
  int wm = wid >> 1, wn = wid & 1;
  int m0 = blockIdx.x * 128 + wm * 32;
  int n0 = blockIdx.y * 128 + wn * 64;

  v8f acc[2][4];
  #pragma unroll
  for (int mi = 0; mi < 2; ++mi)
    #pragma unroll
    for (int ni = 0; ni < 4; ++ni) acc[mi][ni] = zero8();

  for (int k0 = 0; k0 < HIDDEN; k0 += 32) {
    v16bf a0 = load_frag_a(ctx, m0,      k0, HIDDEN);
    v16bf a1 = load_frag_a(ctx, m0 + 16, k0, HIDDEN);
    #pragma unroll
    for (int ni = 0; ni < 4; ++ni) {
      v16bf bfr = load_frag_bt(wo, n0 + ni * 16, k0, HIDDEN);
      acc[0][ni] = wmma_bf16(a0, bfr, acc[0][ni]);
      acc[1][ni] = wmma_bf16(a1, bfr, acc[1][ni]);
    }
  }

  int lane = threadIdx.x & 31, half = lane >> 4, nc = lane & 15;
  #pragma unroll
  for (int mi = 0; mi < 2; ++mi)
    #pragma unroll
    for (int ni = 0; ni < 4; ++ni)
      #pragma unroll
      for (int r = 0; r < 8; ++r) {
        int m = m0 + mi * 16 + half * 8 + r;
        int n = n0 + ni * 16 + nc;
        size_t idx = (size_t)m * HIDDEN + n;
        out[idx] = x[idx] + acc[mi][ni][r] + bo[n];
      }
}

extern "C" void kernel_launch(void* const* d_in, const int* in_sizes, int n_in,
                              void* d_out, int out_size, void* d_ws, size_t ws_size,
                              hipStream_t stream) {
  const float* x    = (const float*)d_in[0];
  const float* Wq   = (const float*)d_in[1];
  const float* bq   = (const float*)d_in[2];
  const float* Wk   = (const float*)d_in[3];
  const float* bk   = (const float*)d_in[4];
  const float* Wv   = (const float*)d_in[5];
  const float* bv   = (const float*)d_in[6];
  const float* Wo   = (const float*)d_in[7];
  const float* bo   = (const float*)d_in[8];
  const float* ln_g = (const float*)d_in[9];
  const float* ln_b = (const float*)d_in[10];

  const size_t MB = 1u << 20;
  char* ws = (char*)d_ws;
  __bf16* xn  = (__bf16*)(ws + 0 * MB);    // 4096x1024 bf16 = 8 MiB
  __bf16* wqb = (__bf16*)(ws + 8 * MB);    // 2 MiB each, contiguous q,k,v,o
  __bf16* wkb = (__bf16*)(ws + 10 * MB);
  __bf16* wvb = (__bf16*)(ws + 12 * MB);
  __bf16* wob = (__bf16*)(ws + 14 * MB);
  __bf16* qb  = (__bf16*)(ws + 16 * MB);   // 8 MiB each
  __bf16* kb  = (__bf16*)(ws + 24 * MB);
  __bf16* vb  = (__bf16*)(ws + 32 * MB);
  __bf16* cb  = (__bf16*)(ws + 40 * MB);   // total 48 MiB

  ln_kernel<<<4096, 256, 0, stream>>>(x, ln_g, ln_b, xn);
  cvt_kernel<<<dim3(1024, 1, 4), 256, 0, stream>>>(Wq, Wk, Wv, Wo, wqb);
  qkv_gemm<<<dim3(32, 8, 3), 256, 0, stream>>>(xn, wqb, wkb, wvb, bq, bk, bv, qb, kb, vb);
  attn_kernel<<<dim3(8, NHEAD, 2), 256, 0, stream>>>(qb, kb, vb, cb);
  oproj_gemm<<<dim3(32, 8, 1), 256, 0, stream>>>(cb, wob, bo, x, (float*)d_out);
}
